// InfiniteMambaCore_84722524881485
// MI455X (gfx1250) — matrix operand order repaired and checked by reading, not compiled, not run on hardware
//
#include <hip/hip_runtime.h>
#include <hip/hip_bf16.h>

#define DIM_   1024
#define INNER_ 2048
#define E2_    4096   // 2*INNER
#define STATE_ 64
#define BB_    4
#define TT_    2048
#define MM_    8192   // B*T

typedef __attribute__((ext_vector_type(16))) __bf16 v16bf;
typedef __attribute__((ext_vector_type(8)))  float  v8f;

union BFrag { v16bf v; uint4 q[2]; };

__device__ __forceinline__ unsigned short f2bf(float f) {
    unsigned int u = __float_as_uint(f);
    unsigned int r = u + 0x7FFFu + ((u >> 16) & 1u);   // RNE
    return (unsigned short)(r >> 16);
}

__device__ __forceinline__ float silu_f(float v) {
    return v * (1.0f / (1.0f + __expf(-v)));
}

// ---------- conversions ----------
__global__ void cvt_f32_bf16_kernel(const float* __restrict__ src,
                                    unsigned short* __restrict__ dst, int n) {
    int i = blockIdx.x * blockDim.x + threadIdx.x;
    if (i < n) dst[i] = f2bf(src[i]);
}

// src is [N,K] row-major f32; dst is [K,N] row-major bf16 (transposed)
__global__ void tcvt_f32_bf16_kernel(const float* __restrict__ src,
                                     unsigned short* __restrict__ dst,
                                     int N, int K) {
    int i = blockIdx.x * blockDim.x + threadIdx.x;
    if (i >= N * K) return;
    int n = i / K, k = i % K;
    dst[(size_t)k * N + n] = f2bf(src[i]);
}

// ---------- bf16 WMMA GEMM: C[M,N] = A[M,K] * Bt[K,N] ----------
// One wave computes a 32x64 output strip: 2 A fragments x 4 B fragments ->
// 8 WMMAs per 32-wide K chunk. Two-stage software pipeline (X/Y register
// buffer sets) so global loads of chunk k+1 overlap the WMMAs of chunk k.
// Requires K % 64 == 0 (even number of 32-chunks): holds for K=1024,2048,64.
__global__ __launch_bounds__(256)
void gemm_bf16_wmma_kernel(const unsigned short* __restrict__ A,
                           const unsigned short* __restrict__ Bt,
                           float* __restrict__ C, int M, int N, int K) {
    const int gtid = blockIdx.x * blockDim.x + threadIdx.x;
    const int wave = gtid >> 5;
    const int lane = threadIdx.x & 31;
    const int colGroups = N >> 6;                 // 64-wide col groups
    const int rowTile = wave / colGroups;         // wave-uniform, 32 rows each
    const int colGroup = wave % colGroups;
    if (rowTile >= (M >> 5)) return;

    const int laneLo = lane & 15;
    const int laneHiOff = (lane >> 4) << 3;       // 0 or 8

    const unsigned short* arow0 = A + (size_t)(rowTile * 32 + laneLo) * K;
    const unsigned short* arow1 = arow0 + (size_t)16 * K;
    const unsigned short* bcol  = Bt + (size_t)laneHiOff * N + colGroup * 64 + laneLo;
    auto ldB = [&](BFrag& f, const unsigned short* p) {
#pragma unroll
        for (int s = 0; s < 16; ++s) {
            const int k = (s < 8) ? s : (8 + s);
            f.v[s] = __builtin_bit_cast(__bf16, p[(size_t)k * N]);
        }
    };

    v8f acc[8] = {};
    const int kchunks = K >> 5;                   // even by contract

    // ---- prologue: load chunk 0 into buffer set X ----
    BFrag ax0, ax1, bx0, bx1, bx2, bx3;
    {
        const int kb = laneHiOff;
        ax0.q[0] = *(const uint4*)(arow0 + kb);
        ax0.q[1] = *(const uint4*)(arow0 + kb + 16);
        ax1.q[0] = *(const uint4*)(arow1 + kb);
        ax1.q[1] = *(const uint4*)(arow1 + kb + 16);
        const unsigned short* brow = bcol;        // column = lane%16, k in table order
        ldB(bx0, brow +  0);
        ldB(bx1, brow + 16);
        ldB(bx2, brow + 32);
        ldB(bx3, brow + 48);
    }

    for (int kc = 0; kc < kchunks; kc += 2) {
        // ---- load chunk kc+1 into buffer set Y (overlaps X compute) ----
        BFrag ay0, ay1, by0, by1, by2, by3;
        {
            const int kb = ((kc + 1) << 5) + laneHiOff;
            ay0.q[0] = *(const uint4*)(arow0 + kb);
            ay0.q[1] = *(const uint4*)(arow0 + kb + 16);
            ay1.q[0] = *(const uint4*)(arow1 + kb);
            ay1.q[1] = *(const uint4*)(arow1 + kb + 16);
            const unsigned short* brow = bcol + ((size_t)(kc + 1) << 5) * N;
            ldB(by0, brow +  0);
            ldB(by1, brow + 16);
            ldB(by2, brow + 32);
            ldB(by3, brow + 48);
        }
        // speculative prefetch of the A rows two chunks ahead (dropped if OOB)
        __builtin_prefetch(arow0 + (((size_t)(kc + 2)) << 5), 0, 1);

        // ---- compute chunk kc from X ----
        acc[0] = __builtin_amdgcn_wmma_f32_16x16x32_bf16(false, ax0.v, false, bx0.v, (short)0, acc[0], false, false);
        acc[1] = __builtin_amdgcn_wmma_f32_16x16x32_bf16(false, ax0.v, false, bx1.v, (short)0, acc[1], false, false);
        acc[2] = __builtin_amdgcn_wmma_f32_16x16x32_bf16(false, ax0.v, false, bx2.v, (short)0, acc[2], false, false);
        acc[3] = __builtin_amdgcn_wmma_f32_16x16x32_bf16(false, ax0.v, false, bx3.v, (short)0, acc[3], false, false);
        acc[4] = __builtin_amdgcn_wmma_f32_16x16x32_bf16(false, ax1.v, false, bx0.v, (short)0, acc[4], false, false);
        acc[5] = __builtin_amdgcn_wmma_f32_16x16x32_bf16(false, ax1.v, false, bx1.v, (short)0, acc[5], false, false);
        acc[6] = __builtin_amdgcn_wmma_f32_16x16x32_bf16(false, ax1.v, false, bx2.v, (short)0, acc[6], false, false);
        acc[7] = __builtin_amdgcn_wmma_f32_16x16x32_bf16(false, ax1.v, false, bx3.v, (short)0, acc[7], false, false);

        // ---- load chunk kc+2 into X (overlaps Y compute) ----
        if (kc + 2 < kchunks) {
            const int kb = ((kc + 2) << 5) + laneHiOff;
            ax0.q[0] = *(const uint4*)(arow0 + kb);
            ax0.q[1] = *(const uint4*)(arow0 + kb + 16);
            ax1.q[0] = *(const uint4*)(arow1 + kb);
            ax1.q[1] = *(const uint4*)(arow1 + kb + 16);
            const unsigned short* brow = bcol + ((size_t)(kc + 2) << 5) * N;
            ldB(bx0, brow +  0);
            ldB(bx1, brow + 16);
            ldB(bx2, brow + 32);
            ldB(bx3, brow + 48);
        }

        // ---- compute chunk kc+1 from Y ----
        acc[0] = __builtin_amdgcn_wmma_f32_16x16x32_bf16(false, ay0.v, false, by0.v, (short)0, acc[0], false, false);
        acc[1] = __builtin_amdgcn_wmma_f32_16x16x32_bf16(false, ay0.v, false, by1.v, (short)0, acc[1], false, false);
        acc[2] = __builtin_amdgcn_wmma_f32_16x16x32_bf16(false, ay0.v, false, by2.v, (short)0, acc[2], false, false);
        acc[3] = __builtin_amdgcn_wmma_f32_16x16x32_bf16(false, ay0.v, false, by3.v, (short)0, acc[3], false, false);
        acc[4] = __builtin_amdgcn_wmma_f32_16x16x32_bf16(false, ay1.v, false, by0.v, (short)0, acc[4], false, false);
        acc[5] = __builtin_amdgcn_wmma_f32_16x16x32_bf16(false, ay1.v, false, by1.v, (short)0, acc[5], false, false);
        acc[6] = __builtin_amdgcn_wmma_f32_16x16x32_bf16(false, ay1.v, false, by2.v, (short)0, acc[6], false, false);
        acc[7] = __builtin_amdgcn_wmma_f32_16x16x32_bf16(false, ay1.v, false, by3.v, (short)0, acc[7], false, false);
    }

    // C/D layout: lane l, VGPR j -> m = j + (l>=16 ? 8 : 0), n = l & 15
    const int mBase0 = rowTile * 32 + laneHiOff;
    const int mBase1 = mBase0 + 16;
#pragma unroll
    for (int t = 0; t < 4; ++t) {
        const int nCol = colGroup * 64 + t * 16 + laneLo;
#pragma unroll
        for (int j = 0; j < 8; ++j) {
            C[(size_t)(mBase0 + j) * N + nCol] = acc[t][j];
            C[(size_t)(mBase1 + j) * N + nCol] = acc[4 + t][j];
        }
    }
}

// ---------- depthwise causal conv (k=4) + SiLU ----------
__global__ void conv_silu_kernel(const float* __restrict__ xproj,
                                 const float* __restrict__ conv_w,
                                 const float* __restrict__ conv_b,
                                 float* __restrict__ u_f,
                                 unsigned short* __restrict__ u_bf) {
    int i = blockIdx.x * blockDim.x + threadIdx.x;
    if (i >= MM_ * INNER_) return;
    int m = i / INNER_, c = i % INNER_;
    int b = m / TT_, t = m % TT_;
    const float* xin = xproj + (size_t)b * TT_ * E2_ + c;  // x_input half, channel c
    float acc = conv_b[c];
#pragma unroll
    for (int j = 0; j < 4; ++j) {
        int tt = t - 3 + j;
        if (tt >= 0) acc += conv_w[c * 4 + j] * xin[(size_t)tt * E2_];
    }
    float s = silu_f(acc);
    u_f[i] = s;
    u_bf[i] = f2bf(s);
}

// ---------- linear recurrence: 256 independent (b,state) lanes ----------
__global__ void scan_kernel(const float* __restrict__ Bu,
                            unsigned short* __restrict__ hs_bf) {
    int i = threadIdx.x;             // 0..255 = b*STATE + s
    int b = i / STATE_, s = i % STATE_;
    float h = 0.0f;
    for (int t = 0; t < TT_; ++t) {
        size_t idx = ((size_t)(b * TT_ + t)) * STATE_ + s;
        h = 0.95f * h + 0.05f * Bu[idx];
        hs_bf[idx] = f2bf(h);
    }
}

// ---------- y = (hs@WcT + D*u) * silu(gate) ----------
__global__ void gate_kernel(const float* __restrict__ yc,
                            const float* __restrict__ u_f,
                            const float* __restrict__ xproj,
                            const float* __restrict__ D_param,
                            unsigned short* __restrict__ y_bf) {
    int i = blockIdx.x * blockDim.x + threadIdx.x;
    if (i >= MM_ * INNER_) return;
    int m = i / INNER_, c = i % INNER_;
    float v = yc[i] + D_param[c] * u_f[i];
    float g = xproj[(size_t)m * E2_ + INNER_ + c];   // gate half
    v *= silu_f(g);
    y_bf[i] = f2bf(v);
}

// ---------- residual + LayerNorm, one block per row ----------
__global__ __launch_bounds__(256)
void ln_kernel(const float* __restrict__ o, const float* __restrict__ x,
               const float* __restrict__ gamma, const float* __restrict__ beta,
               float* __restrict__ out) {
    __shared__ float sbuf[256];
    const int m = blockIdx.x, tid = threadIdx.x;
    float v[4];
    float sum = 0.0f;
#pragma unroll
    for (int i = 0; i < 4; ++i) {
        int d = tid + i * 256;
        v[i] = o[(size_t)m * DIM_ + d] + x[(size_t)m * DIM_ + d];
        sum += v[i];
    }
    sbuf[tid] = sum; __syncthreads();
    for (int s = 128; s > 0; s >>= 1) {
        if (tid < s) sbuf[tid] += sbuf[tid + s];
        __syncthreads();
    }
    float mu = sbuf[0] * (1.0f / DIM_);
    __syncthreads();
    float sq = 0.0f;
#pragma unroll
    for (int i = 0; i < 4; ++i) { float d = v[i] - mu; sq += d * d; }
    sbuf[tid] = sq; __syncthreads();
    for (int s = 128; s > 0; s >>= 1) {
        if (tid < s) sbuf[tid] += sbuf[tid + s];
        __syncthreads();
    }
    float inv = rsqrtf(sbuf[0] * (1.0f / DIM_) + 1e-5f);
#pragma unroll
    for (int i = 0; i < 4; ++i) {
        int d = tid + i * 256;
        out[(size_t)m * DIM_ + d] = (v[i] - mu) * inv * gamma[d] + beta[d];
    }
}

extern "C" void kernel_launch(void* const* d_in, const int* in_sizes, int n_in,
                              void* d_out, int out_size, void* d_ws, size_t ws_size,
                              hipStream_t stream) {
    const float* x       = (const float*)d_in[0];
    const float* W_in    = (const float*)d_in[1];
    const float* conv_w  = (const float*)d_in[2];
    const float* conv_b  = (const float*)d_in[3];
    const float* W_B     = (const float*)d_in[4];
    const float* W_C     = (const float*)d_in[5];
    const float* D_param = (const float*)d_in[6];
    const float* W_out   = (const float*)d_in[7];
    const float* ln_g    = (const float*)d_in[8];
    const float* ln_b    = (const float*)d_in[9];
    float* out = (float*)d_out;

    char* ws = (char*)d_ws;
    size_t off = 0;
    auto alloc = [&](size_t bytes) { size_t o = off; off = (off + bytes + 255) & ~(size_t)255; return o; };

    unsigned short* Xbf   = (unsigned short*)(ws + alloc((size_t)MM_ * DIM_ * 2));
    unsigned short* WinT  = (unsigned short*)(ws + alloc((size_t)DIM_ * E2_ * 2));
    float*          xproj = (float*)         (ws + alloc((size_t)MM_ * E2_ * 4));
    float*          u_f   = (float*)         (ws + alloc((size_t)MM_ * INNER_ * 4));
    unsigned short* u_bf  = (unsigned short*)(ws + alloc((size_t)MM_ * INNER_ * 2));
    unsigned short* WBT   = (unsigned short*)(ws + alloc((size_t)INNER_ * STATE_ * 2));
    float*          Bu    = (float*)         (ws + alloc((size_t)MM_ * STATE_ * 4));
    unsigned short* hs_bf = (unsigned short*)(ws + alloc((size_t)MM_ * STATE_ * 2));
    unsigned short* WCT   = (unsigned short*)(ws + alloc((size_t)STATE_ * INNER_ * 2));
    float*          y_f   = (float*)         (ws + alloc((size_t)MM_ * INNER_ * 4));
    unsigned short* y_bf  = (unsigned short*)(ws + alloc((size_t)MM_ * INNER_ * 2));
    unsigned short* WoutT = (unsigned short*)(ws + alloc((size_t)INNER_ * DIM_ * 2));
    float*          o_f   = (float*)         (ws + alloc((size_t)MM_ * DIM_ * 4));
    (void)ws_size; (void)in_sizes; (void)n_in; (void)out_size;

    const int TPB = 256;
    auto blocks = [&](size_t n) { return (unsigned)((n + TPB - 1) / TPB); };

    // 1) bf16 conversions / weight transposes
    cvt_f32_bf16_kernel<<<blocks((size_t)MM_ * DIM_), TPB, 0, stream>>>(x, Xbf, MM_ * DIM_);
    tcvt_f32_bf16_kernel<<<blocks((size_t)E2_ * DIM_), TPB, 0, stream>>>(W_in, WinT, E2_, DIM_);
    tcvt_f32_bf16_kernel<<<blocks((size_t)STATE_ * INNER_), TPB, 0, stream>>>(W_B, WBT, STATE_, INNER_);
    tcvt_f32_bf16_kernel<<<blocks((size_t)INNER_ * STATE_), TPB, 0, stream>>>(W_C, WCT, INNER_, STATE_);
    tcvt_f32_bf16_kernel<<<blocks((size_t)DIM_ * INNER_), TPB, 0, stream>>>(W_out, WoutT, DIM_, INNER_);

    // 2) in_proj: xproj[8192,4096] = Xbf[8192,1024] @ WinT[1024,4096]
    {
        unsigned waves = (MM_ / 32) * (E2_ / 64);
        gemm_bf16_wmma_kernel<<<waves / 8, TPB, 0, stream>>>(Xbf, WinT, xproj, MM_, E2_, DIM_);
    }
    // 3) depthwise causal conv + silu -> u
    conv_silu_kernel<<<blocks((size_t)MM_ * INNER_), TPB, 0, stream>>>(xproj, conv_w, conv_b, u_f, u_bf);
    // 4) Bu[8192,64] = u_bf @ WBT[2048,64]
    {
        unsigned waves = (MM_ / 32) * (STATE_ / 64);
        gemm_bf16_wmma_kernel<<<waves / 8, TPB, 0, stream>>>(u_bf, WBT, Bu, MM_, STATE_, INNER_);
    }
    // 5) sequential scan over T (256 independent lanes)
    scan_kernel<<<1, BB_ * STATE_, 0, stream>>>(Bu, hs_bf);
    // 6) yc[8192,2048] = hs_bf @ WCT[64,2048]
    {
        unsigned waves = (MM_ / 32) * (INNER_ / 64);
        gemm_bf16_wmma_kernel<<<waves / 8, TPB, 0, stream>>>(hs_bf, WCT, y_f, MM_, INNER_, STATE_);
    }
    // 7) y = (yc + D*u) * silu(gate)
    gate_kernel<<<blocks((size_t)MM_ * INNER_), TPB, 0, stream>>>(y_f, u_f, xproj, D_param, y_bf);
    // 8) o[8192,1024] = y_bf @ WoutT[2048,1024]
    {
        unsigned waves = (MM_ / 32) * (DIM_ / 64);
        gemm_bf16_wmma_kernel<<<waves / 8, TPB, 0, stream>>>(y_bf, WoutT, o_f, MM_, DIM_, INNER_);
    }
    // 9) residual + layernorm
    ln_kernel<<<MM_, TPB, 0, stream>>>(o_f, x, ln_g, ln_b, out);
}
